// EHRHierarchicalLoss_51307679318817
// MI455X (gfx1250) — matrix operand order repaired and checked by reading, not compiled, not run on hardware
//
#include <hip/hip_runtime.h>
#include <hip/hip_bf16.h>

// ---------------------------------------------------------------------------
// EHR hierarchical contrastive loss for MI455X (gfx1250, wave32, WMMA).
//   sim = (norm(R) @ norm(R)^T) / 0.1          -> v_wmma_f32_16x16x32_f16
//   Jaccard positives via 128-bit popcount masks (VOCAB=100 -> 4 x u32)
//   pass1: neg_sum, n_pos   pass2: sum_pos[log(exp(s)+neg_sum)-s]
// Deterministic: per-block partials + fixed-order tree reductions, no atomics.
// ---------------------------------------------------------------------------

typedef __attribute__((ext_vector_type(16))) _Float16 v16h;
typedef __attribute__((ext_vector_type(8)))  float    v8f;

#define B_N 4096
#define D_N 256
#define M_N 60
#define TEMP_INV 10.0f   // 1 / TEMPERATURE

// ---- helpers --------------------------------------------------------------

__device__ __forceinline__ v16h load16h(const _Float16* p0, const _Float16* p1) {
  // two 16-byte loads -> one 8-VGPR f16 fragment (global_load_b128 x2)
  union { uint4 u[2]; v16h h; } t;
  t.u[0] = *(const uint4*)p0;
  t.u[1] = *(const uint4*)p1;
  return t.h;
}

__device__ __forceinline__ float block_reduce(float v, float* sRed, int tid) {
  sRed[tid] = v;
  __syncthreads();
  #pragma unroll
  for (int off = 128; off > 0; off >>= 1) {
    if (tid < off) sRed[tid] += sRed[tid + off];
    __syncthreads();
  }
  float r = sRed[0];
  __syncthreads();   // safe for buffer reuse
  return r;
}

// ---- 1) normalize rows, emit f16 copy -------------------------------------

__global__ __launch_bounds__(256) void normalize_kernel(
    const float* __restrict__ reps, _Float16* __restrict__ nrm) {
  __shared__ float sRed[256];
  const int row = blockIdx.x;
  const int tid = threadIdx.x;
  const float x = reps[(size_t)row * D_N + tid];
  const float ss = block_reduce(x * x, sRed, tid);
  const float inv = rsqrtf(ss);
  nrm[(size_t)row * D_N + tid] = (_Float16)(x * inv);
}

// ---- 2) build 128-bit membership masks + unique-code counts ---------------

__global__ __launch_bounds__(256) void build_masks_kernel(
    const int* __restrict__ codes, uint4* __restrict__ masks,
    unsigned* __restrict__ cnts) {
  const int row = blockIdx.x * 256 + threadIdx.x;
  if (row >= B_N) return;
  unsigned w0 = 0, w1 = 0, w2 = 0, w3 = 0;
  const int* c = codes + (size_t)row * M_N;
  #pragma unroll 4
  for (int i = 0; i < M_N; ++i) {
    const unsigned code = (unsigned)c[i];       // in [0, 100)
    const unsigned bit  = 1u << (code & 31u);
    const unsigned w    = code >> 5;
    if (w == 0) w0 |= bit; else if (w == 1) w1 |= bit;
    else if (w == 2) w2 |= bit; else w3 |= bit;
  }
  masks[row] = make_uint4(w0, w1, w2, w3);
  cnts[row]  = (unsigned)(__popc(w0) + __popc(w1) + __popc(w2) + __popc(w3));
}

// ---- 3/5) tiled WMMA similarity passes ------------------------------------
// Block: 256 thr = 8 waves, 128x128 tile. Wave (2x4 grid) owns 64x32.
// PASS 1: partial neg_sum (non-positives incl. diagonal) + partial n_pos.
// PASS 2: partial positive loss using scalar neg_sum.

template <int PASS>
__global__ __launch_bounds__(256) void pass_kernel(
    const _Float16* __restrict__ Ah,
    const uint4* __restrict__ masks, const unsigned* __restrict__ cnts,
    const float* __restrict__ scalars,
    float* __restrict__ outA, float* __restrict__ outB) {
  __shared__ uint4    sMaskR[128];
  __shared__ unsigned sCntR[128];
  __shared__ uint4    sMaskC[128];
  __shared__ unsigned sCntC[128];
  __shared__ float    sRed[256];

  const int tid    = threadIdx.x;
  const int lane   = tid & 31;
  const int wave   = tid >> 5;
  const int waveM  = wave >> 2;          // 0..1
  const int waveN  = wave & 3;           // 0..3
  const int blockM = blockIdx.y * 128;
  const int blockN = blockIdx.x * 128;

  if (tid < 128) { sMaskR[tid] = masks[blockM + tid]; sCntR[tid] = cnts[blockM + tid]; }
  else { const int t = tid - 128; sMaskC[t] = masks[blockN + t]; sCntC[t] = cnts[blockN + t]; }
  __syncthreads();

  const int hi  = (lane < 16) ? 0 : 1;   // half-wave selector per ISA layouts
  const int lm  = lane & 15;
  const int row0 = blockM + waveM * 64 + lm;   // A fragment rows (+ i*16)
  const int col0 = blockN + waveN * 32 + lm;   // B fragment cols = rows of Ah (+ j*16)

  v8f acc[4][2];
  #pragma unroll
  for (int i = 0; i < 4; ++i)
    #pragma unroll
    for (int j = 0; j < 2; ++j) acc[i][j] = v8f{};

  #pragma unroll 2
  for (int k0 = 0; k0 < D_N; k0 += 32) {
    // A (16x32 f16): lane<16 holds K 0..7 & 16..23, lane>=16 holds 8..15 & 24..31
    const int ka = k0 + (hi ? 8 : 0);
    v16h a[4];
    #pragma unroll
    for (int i = 0; i < 4; ++i) {
      const _Float16* pa = Ah + (size_t)(row0 + i * 16) * D_N + ka;
      a[i] = load16h(pa, pa + 16);
    }
    // B (32x16 f16): lane = column N, lane<16 holds K 0..7 & 16..23, lane>=16 holds 8..15 & 24..31 (B = Ah^T slice)
    const int kb = k0 + (hi ? 8 : 0);
    v16h b[2];
    #pragma unroll
    for (int j = 0; j < 2; ++j) {
      const _Float16* pb = Ah + (size_t)(col0 + j * 16) * D_N + kb;
      b[j] = load16h(pb, pb + 16);
    }
    #pragma unroll
    for (int i = 0; i < 4; ++i)
      #pragma unroll
      for (int j = 0; j < 2; ++j)
        acc[i][j] = __builtin_amdgcn_wmma_f32_16x16x32_f16(
            false, a[i], false, b[j], (short)0, acc[i][j], false, false);
  }

  // Epilogue: C/D layout -> (m,n); Jaccard via popcount; accumulate.
  const float negSum = (PASS == 2) ? scalars[0] : 0.0f;
  float localA = 0.0f;   // PASS1: neg_sum partial | PASS2: loss partial
  float localB = 0.0f;   // PASS1: n_pos partial

  #pragma unroll
  for (int i = 0; i < 4; ++i) {
    #pragma unroll
    for (int j = 0; j < 2; ++j) {
      const int liC = waveN * 32 + j * 16 + lm;
      const uint4    cm = sMaskC[liC];
      const unsigned cc = sCntC[liC];
      const int      gn = blockN + liC;
      #pragma unroll
      for (int r = 0; r < 8; ++r) {
        const int liR = waveM * 64 + i * 16 + (hi ? r + 8 : r);
        const uint4    rm = sMaskR[liR];
        const unsigned rc = sCntR[liR];
        const int      gm = blockM + liR;
        const unsigned inter = (unsigned)(__popc(rm.x & cm.x) + __popc(rm.y & cm.y) +
                                          __popc(rm.z & cm.z) + __popc(rm.w & cm.w));
        const unsigned uni = rc + cc - inter;
        const bool pos = (10u * inter > 3u * uni) && (gm != gn);  // jacc > 0.3, off-diag
        const float s = acc[i][j][r] * TEMP_INV;
        if (PASS == 1) {
          if (pos) localB += 1.0f;
          else     localA += __expf(s);      // negatives incl. diagonal
        } else {
          if (pos) localA += __logf(__expf(s) + negSum) - s;
        }
      }
    }
  }

  const int bid = blockIdx.y * 32 + blockIdx.x;
  const float totA = block_reduce(localA, sRed, tid);
  if (PASS == 1) {
    const float totB = block_reduce(localB, sRed, tid);
    if (tid == 0) { outA[bid] = totA; outB[bid] = totB; }
  } else {
    if (tid == 0) outA[bid] = totA;
  }
}

// ---- 4) deterministic reduction of pass-1 partials ------------------------

__global__ __launch_bounds__(256) void reduce1_kernel(
    const float* __restrict__ negPart, const float* __restrict__ npPart,
    float* __restrict__ scalars) {
  __shared__ float sRed[256];
  const int tid = threadIdx.x;
  float a = negPart[tid] + negPart[tid + 256] + negPart[tid + 512] + negPart[tid + 768];
  float b = npPart[tid]  + npPart[tid + 256]  + npPart[tid + 512]  + npPart[tid + 768];
  const float negSum = block_reduce(a, sRed, tid);
  const float nPos   = block_reduce(b, sRed, tid);
  if (tid == 0) { scalars[0] = negSum; scalars[1] = nPos; }
}

// ---- 6) finalize ----------------------------------------------------------

__global__ __launch_bounds__(256) void finalize_kernel(
    const float* __restrict__ lossPart, const float* __restrict__ scalars,
    float* __restrict__ out) {
  __shared__ float sRed[256];
  const int tid = threadIdx.x;
  float a = lossPart[tid] + lossPart[tid + 256] + lossPart[tid + 512] + lossPart[tid + 768];
  const float L = block_reduce(a, sRed, tid);
  if (tid == 0) {
    const float nPos = scalars[1];
    out[0] = (nPos > 0.0f) ? (L / fmaxf(nPos, 1.0f)) : 0.0f;
  }
}

// ---- launch ---------------------------------------------------------------

extern "C" void kernel_launch(void* const* d_in, const int* in_sizes, int n_in,
                              void* d_out, int out_size, void* d_ws, size_t ws_size,
                              hipStream_t stream) {
  const float* reps  = (const float*)d_in[0];
  const int*   codes = (const int*)d_in[1];   // medical_codes (int), labels d_in[2] unused
  float* out = (float*)d_out;

  char* ws = (char*)d_ws;
  float*     negPart  = (float*)(ws + 0);                 // 1024 f32
  float*     npPart   = (float*)(ws + 4096);              // 1024 f32
  float*     lossPart = (float*)(ws + 8192);              // 1024 f32
  float*     scalars  = (float*)(ws + 12288);             // [neg_sum, n_pos]
  _Float16*  nrm      = (_Float16*)(ws + 16384);          // 4096x256 f16 (2 MB)
  uint4*     masks    = (uint4*)(ws + 16384 + 2097152);   // 4096 x uint4
  unsigned*  cnts     = (unsigned*)(ws + 16384 + 2097152 + 65536);

  normalize_kernel<<<B_N, 256, 0, stream>>>(reps, nrm);
  build_masks_kernel<<<B_N / 256, 256, 0, stream>>>(codes, masks, cnts);

  dim3 grid(32, 32, 1);
  pass_kernel<1><<<grid, 256, 0, stream>>>(nrm, masks, cnts, scalars, negPart, npPart);
  reduce1_kernel<<<1, 256, 0, stream>>>(negPart, npPart, scalars);
  pass_kernel<2><<<grid, 256, 0, stream>>>(nrm, masks, cnts, scalars, lossPart, nullptr);
  finalize_kernel<<<1, 256, 0, stream>>>(lossPart, scalars, out);
}
